// LinearCentroids_27462020891406
// MI455X (gfx1250) — matrix operand order repaired and checked by reading, not compiled, not run on hardware
//
#include <hip/hip_runtime.h>
#include <hip/hip_bf16.h>
#include <math.h>

typedef __attribute__((ext_vector_type(16))) __bf16 v16bf;
typedef __attribute__((ext_vector_type(8)))  __bf16 v8bf;
typedef __attribute__((ext_vector_type(8)))  float  v8f;
typedef __attribute__((ext_vector_type(4)))  int    v4i;

#define B_DIM 8192
#define E_DIM 2048
#define F_DIM 128
#define C_DIM 1000
#define C_PAD 1024

// ---------------------------------------------------------------------------
// CDNA5 async global->LDS copy (ASYNCcnt path) with safe fallback
// Probe-confirmed prototype: (v4i AS1* gsrc, v4i AS3* lds, Ii offset, Ii cpol)
// ---------------------------------------------------------------------------
#if defined(__AMDGCN__) && __has_builtin(__builtin_amdgcn_global_load_async_to_lds_b128)
#define USE_ASYNC_LDS 1
#endif

__device__ __forceinline__ void copy16_g2s(__bf16* lds_dst, const __bf16* gsrc) {
#ifdef USE_ASYNC_LDS
  __builtin_amdgcn_global_load_async_to_lds_b128(
      (__attribute__((address_space(1))) v4i*)(const_cast<__bf16*>(gsrc)),
      (__attribute__((address_space(3))) v4i*)(lds_dst),
      /*offset=*/0, /*cpol=*/0);
#else
  *reinterpret_cast<v8bf*>(lds_dst) = *reinterpret_cast<const v8bf*>(gsrc);
#endif
}

__device__ __forceinline__ void wait_async_copies() {
#ifdef USE_ASYNC_LDS
#if __has_builtin(__builtin_amdgcn_s_wait_asynccnt)
  __builtin_amdgcn_s_wait_asynccnt(0);
#else
  asm volatile("s_wait_asynccnt 0" ::: "memory");
#endif
#endif
}

// ---------------------------------------------------------------------------
// helpers
// ---------------------------------------------------------------------------
__device__ __forceinline__ float wave_reduce_sum(float v) {
#pragma unroll
  for (int m = 16; m; m >>= 1) v += __shfl_xor(v, m, 32);
  return v;
}

__device__ __forceinline__ void split_bf16(float x, __bf16& hi, __bf16& lo) {
  __bf16 h = (__bf16)x;            // RNE
  hi = h;
  lo = (__bf16)(x - (float)h);     // residual -> ~16 mantissa bits total
}

__device__ __forceinline__ v8f wmma3(v16bf ahi, v16bf alo, v16bf bhi, v16bf blo,
                                     v8f c) {
  c = __builtin_amdgcn_wmma_f32_16x16x32_bf16(false, alo, false, bhi, (short)0, c,
                                              false, false);
  c = __builtin_amdgcn_wmma_f32_16x16x32_bf16(false, ahi, false, blo, (short)0, c,
                                              false, false);
  c = __builtin_amdgcn_wmma_f32_16x16x32_bf16(false, ahi, false, bhi, (short)0, c,
                                              false, false);
  return c;
}

// ---------------------------------------------------------------------------
// 1) spectral sigma: inv_sigma = 1 / (||W v_hat|| + eps), v_hat = norm(W^T u)
//    single block, 1024 threads (32 waves)
// ---------------------------------------------------------------------------
__global__ void sigma_kernel(const float* __restrict__ W,
                             const float* __restrict__ u,
                             float* __restrict__ inv_sigma) {
  __shared__ float s_u[F_DIM];
  __shared__ float s_v[E_DIM];
  __shared__ float s_t[F_DIM];
  __shared__ float s_red[32];
  const int t = threadIdx.x;
  const int lane = t & 31, wid = t >> 5;

  if (t < F_DIM) s_u[t] = u[t];
  __syncthreads();

  // v = W^T u   (each thread: 2 columns)
  for (int j = t; j < E_DIM; j += 1024) {
    float acc = 0.f;
#pragma unroll 8
    for (int i = 0; i < F_DIM; ++i) acc += W[(size_t)i * E_DIM + j] * s_u[i];
    s_v[j] = acc;
  }
  __syncthreads();

  // ||v||
  float ss = 0.f;
  for (int j = t; j < E_DIM; j += 1024) { float v = s_v[j]; ss += v * v; }
  ss = wave_reduce_sum(ss);
  if (lane == 0) s_red[wid] = ss;
  __syncthreads();
  if (wid == 0) {
    float r = s_red[lane];
    r = wave_reduce_sum(r);
    if (lane == 0) s_red[0] = r;
  }
  __syncthreads();
  const float inv_vn = 1.0f / (sqrtf(s_red[0]) + 1e-12f);
  for (int j = t; j < E_DIM; j += 1024) s_v[j] *= inv_vn;
  __syncthreads();

  // t = W v_hat  (8 threads per row, aligned inside a wave)
  {
    const int row = t >> 3, sub = t & 7;
    float acc = 0.f;
    for (int j = sub; j < E_DIM; j += 8) acc += W[(size_t)row * E_DIM + j] * s_v[j];
    acc += __shfl_xor(acc, 1, 32);
    acc += __shfl_xor(acc, 2, 32);
    acc += __shfl_xor(acc, 4, 32);
    if (sub == 0) s_t[row] = acc;
  }
  __syncthreads();

  // sigma ~= ||t||
  if (t < 32) {
    float a = 0.f;
#pragma unroll
    for (int i = t; i < F_DIM; i += 32) { float x = s_t[i]; a += x * x; }
    a = wave_reduce_sum(a);
    if (t == 0) *inv_sigma = 1.0f / (sqrtf(a) + 1e-12f);
  }
}

// ---------------------------------------------------------------------------
// 2) split W (f32) -> bf16 hi/lo (row-major, same indexing)
// ---------------------------------------------------------------------------
__global__ void split_w_kernel(const float* __restrict__ W,
                               __bf16* __restrict__ hi, __bf16* __restrict__ lo) {
  const int i = blockIdx.x * blockDim.x + threadIdx.x;  // exact grid
  const float x = W[i];
  __bf16 h, l;
  split_bf16(x, h, l);
  hi[i] = h; lo[i] = l;
}

// ---------------------------------------------------------------------------
// 3) split c -> bf16 hi/lo (padded to C_PAD rows of zeros) + row norms
//    one wave per row
// ---------------------------------------------------------------------------
__global__ void prep_c_kernel(const float* __restrict__ c,
                              __bf16* __restrict__ chi, __bf16* __restrict__ clo,
                              float* __restrict__ cnorm) {
  const int row  = (blockIdx.x * blockDim.x + threadIdx.x) >> 5;
  const int lane = threadIdx.x & 31;
  float ss = 0.f;
#pragma unroll
  for (int k = lane; k < F_DIM; k += 32) {
    const float x = (row < C_DIM) ? c[(size_t)row * F_DIM + k] : 0.0f;
    __bf16 h, l;
    split_bf16(x, h, l);
    chi[(size_t)row * F_DIM + k] = h;
    clo[(size_t)row * F_DIM + k] = l;
    ss += x * x;
  }
  ss = wave_reduce_sum(ss);
  if (lane == 0) cnorm[row] = ss;
}

// ---------------------------------------------------------------------------
// A-fragment builder: 16x32 bf16 A tile in contiguous K order (k = 16*half + e), same as B.
// lane(0..15)=row, half selects contiguous K chunk {half*16..+15}
// ---------------------------------------------------------------------------
__device__ __forceinline__ void load_split_A(const float* __restrict__ rowp,
                                             int k0, int half,
                                             v16bf& ahi, v16bf& alo) {
  float af[16];
  const float4* p0 = reinterpret_cast<const float4*>(rowp + k0 + half * 16);
  const float4* p1 = reinterpret_cast<const float4*>(rowp + k0 + half * 16 + 8);
  float4 q0 = p0[0], q1 = p0[1], q2 = p1[0], q3 = p1[1];
  af[0] = q0.x; af[1] = q0.y; af[2]  = q0.z; af[3]  = q0.w;
  af[4] = q1.x; af[5] = q1.y; af[6]  = q1.z; af[7]  = q1.w;
  af[8] = q2.x; af[9] = q2.y; af[10] = q2.z; af[11] = q2.w;
  af[12] = q3.x; af[13] = q3.y; af[14] = q3.z; af[15] = q3.w;
#pragma unroll
  for (int e = 0; e < 16; ++e) {
    __bf16 h, l;
    split_bf16(af[e], h, l);
    ahi[e] = h; alo[e] = l;
  }
}

// ---------------------------------------------------------------------------
// 4) GEMM1: z = (x @ W^T) * inv_sigma + b      [8192 x 128]
//    block = 256 thr (8 waves); block tile 128M x 128N; wave tile 16M x 128N.
//    W hi/lo k-tiles staged in LDS (double-buffered, async copies):
//    8x less L2 traffic than per-wave global B loads.
// ---------------------------------------------------------------------------
__global__ void __launch_bounds__(256)
gemm1_kernel(const float* __restrict__ x,
             const __bf16* __restrict__ Whi, const __bf16* __restrict__ Wlo,
             const float* __restrict__ bias, const float* __restrict__ inv_sigma_p,
             float* __restrict__ z) {
  __shared__ __align__(64) __bf16 sW[2][2][F_DIM * 32];  // [buf][hi/lo], 32 KB
  const int tid  = threadIdx.x;
  const int wave = tid >> 5;
  const int lane = tid & 31;
  const int half = lane >> 4;
  const int l16  = lane & 15;
  const int mbase = blockIdx.x * 128 + wave * 16;

  // cooperative copy of one 128x32 k-tile of Whi+Wlo into LDS buffer `buf`
  auto issue_copy = [&](int buf, int k0) {
#pragma unroll
    for (int cpy = 0; cpy < 2; ++cpy) {
      const int chunk = cpy * 256 + tid;       // 0..511 chunks of 8 bf16
      const int row   = chunk >> 2;            // 4 chunks per 32-elem row
      const int col   = (chunk & 3) * 8;
      const size_t goff = (size_t)row * E_DIM + k0 + col;
      const int    loff = row * 32 + col;
      copy16_g2s(&sW[buf][0][loff], Whi + goff);
      copy16_g2s(&sW[buf][1][loff], Wlo + goff);
    }
  };

  v8f acc[8] = {};
  const float* xrow = x + (size_t)(mbase + l16) * E_DIM;

  issue_copy(0, 0);

  for (int ks = 0; ks < E_DIM / 32; ++ks) {
    const int k0  = ks * 32;
    const int cur = ks & 1;
    wait_async_copies();     // my copies done (incl. those landing in sW[cur])
    __syncthreads();         // everyone's copies visible; prev buf reads done
    if (ks + 1 < E_DIM / 32) issue_copy(cur ^ 1, k0 + 32);

    __builtin_prefetch(xrow + k0 + 128, 0, 0);   // stream next x chunk
    v16bf ahi, alo;
    load_split_A(xrow, k0, half, ahi, alo);
#pragma unroll
    for (int n = 0; n < 8; ++n) {
      // B 32x16: lane col = l16, K = half*16 + [0..15]  (32B contiguous in LDS)
      const int boff = (n * 16 + l16) * 32 + half * 16;
      v16bf bhi = *reinterpret_cast<const v16bf*>(&sW[cur][0][boff]);
      v16bf blo = *reinterpret_cast<const v16bf*>(&sW[cur][1][boff]);
      acc[n] = wmma3(ahi, alo, bhi, blo, acc[n]);
    }
  }

  const float is = *inv_sigma_p;
  // C/D layout: VGPR r -> M = r + 8*half ; N = l16
#pragma unroll
  for (int n = 0; n < 8; ++n) {
    const int col = n * 16 + l16;
    const float bc = bias[col];
#pragma unroll
    for (int r = 0; r < 8; ++r) {
      const int row = mbase + r + 8 * half;
      z[(size_t)row * F_DIM + col] = acc[n][r] * is + bc;
    }
  }
}

// ---------------------------------------------------------------------------
// 5) row norms of z: one wave per row
// ---------------------------------------------------------------------------
__global__ void znorm_kernel(const float* __restrict__ z, float* __restrict__ zn) {
  const int row  = (blockIdx.x * blockDim.x + threadIdx.x) >> 5;
  const int lane = threadIdx.x & 31;
  float ss = 0.f;
#pragma unroll
  for (int k = lane; k < F_DIM; k += 32) {
    const float v = z[(size_t)row * F_DIM + k];
    ss += v * v;
  }
  ss = wave_reduce_sum(ss);
  if (lane == 0) zn[row] = ss;
}

// ---------------------------------------------------------------------------
// 6) GEMM2 + epilogue: out = exp(-(||z||^2 + ||c||^2 - 2 z.c)/2)
//    grid = (M/128, C_PAD/128); whole 128x128 hi/lo c tile staged in LDS once.
// ---------------------------------------------------------------------------
__global__ void __launch_bounds__(256)
gemm2_kernel(const float* __restrict__ z,
             const __bf16* __restrict__ chi, const __bf16* __restrict__ clo,
             const float* __restrict__ zn, const float* __restrict__ cn,
             float* __restrict__ out) {
  __shared__ __align__(64) __bf16 sC[2][128 * F_DIM];    // hi/lo planes, 64 KB
  const int tid  = threadIdx.x;
  const int wave = tid >> 5;
  const int lane = tid & 31;
  const int half = lane >> 4;
  const int l16  = lane & 15;
  const int mbase = blockIdx.x * 128 + wave * 16;
  const int nbase = blockIdx.y * 128;

  // stage entire c tile: 2048 chunks of 8 bf16 per plane, 8 per thread
#pragma unroll
  for (int cpy = 0; cpy < 8; ++cpy) {
    const int chunk = cpy * 256 + tid;       // 0..2047
    const int row   = chunk >> 4;            // 16 chunks per 128-elem row
    const int col   = (chunk & 15) * 8;
    const size_t goff = (size_t)(nbase + row) * F_DIM + col;
    const int    loff = row * F_DIM + col;
    copy16_g2s(&sC[0][loff], chi + goff);
    copy16_g2s(&sC[1][loff], clo + goff);
  }
  wait_async_copies();
  __syncthreads();

  v8f acc[8] = {};
  const float* zrow = z + (size_t)(mbase + l16) * F_DIM;

#pragma unroll
  for (int ks = 0; ks < F_DIM / 32; ++ks) {
    const int k0 = ks * 32;
    v16bf ahi, alo;
    load_split_A(zrow, k0, half, ahi, alo);
#pragma unroll
    for (int n = 0; n < 8; ++n) {
      const int boff = (n * 16 + l16) * F_DIM + k0 + half * 16;
      v16bf bhi = *reinterpret_cast<const v16bf*>(&sC[0][boff]);
      v16bf blo = *reinterpret_cast<const v16bf*>(&sC[1][boff]);
      acc[n] = wmma3(ahi, alo, bhi, blo, acc[n]);
    }
  }

#pragma unroll
  for (int n = 0; n < 8; ++n) {
    const int col = nbase + n * 16 + l16;
    if (col < C_DIM) {
      const float cnv = cn[col];
#pragma unroll
      for (int r = 0; r < 8; ++r) {
        const int row = mbase + r + 8 * half;
        float sq = zn[row] + cnv - 2.0f * acc[n][r];
        sq = fmaxf(sq, 0.0f);
        out[(size_t)row * C_DIM + col] = __expf(-0.5f * sq);
      }
    }
  }
}

// ---------------------------------------------------------------------------
// launch
// ---------------------------------------------------------------------------
extern "C" void kernel_launch(void* const* d_in, const int* in_sizes, int n_in,
                              void* d_out, int out_size, void* d_ws, size_t ws_size,
                              hipStream_t stream) {
  const float* x = (const float*)d_in[0];   // [8192, 2048]
  const float* W = (const float*)d_in[1];   // [128, 2048]
  const float* b = (const float*)d_in[2];   // [128]
  const float* u = (const float*)d_in[3];   // [128]
  const float* c = (const float*)d_in[4];   // [1000, 128]
  float* out = (float*)d_out;               // [8192, 1000]

  // workspace carve-out (256B-aligned chunks)
  char* ws = (char*)d_ws;
  auto carve = [&](size_t bytes) -> char* {
    char* p = ws;
    ws += (bytes + 255) & ~(size_t)255;
    return p;
  };
  float*  inv_sigma = (float*)  carve(sizeof(float));
  __bf16* Whi       = (__bf16*) carve((size_t)F_DIM * E_DIM * 2);
  __bf16* Wlo       = (__bf16*) carve((size_t)F_DIM * E_DIM * 2);
  __bf16* chi       = (__bf16*) carve((size_t)C_PAD * F_DIM * 2);
  __bf16* clo       = (__bf16*) carve((size_t)C_PAD * F_DIM * 2);
  float*  cnorm     = (float*)  carve((size_t)C_PAD * 4);
  float*  zbuf      = (float*)  carve((size_t)B_DIM * F_DIM * 4);
  float*  znorm     = (float*)  carve((size_t)B_DIM * 4);

  sigma_kernel<<<1, 1024, 0, stream>>>(W, u, inv_sigma);
  split_w_kernel<<<(F_DIM * E_DIM) / 256, 256, 0, stream>>>(W, Whi, Wlo);
  prep_c_kernel<<<(C_PAD * 32) / 256, 256, 0, stream>>>(c, chi, clo, cnorm);
  gemm1_kernel<<<B_DIM / 128, 256, 0, stream>>>(x, Whi, Wlo, b, inv_sigma, zbuf);
  znorm_kernel<<<(B_DIM * 32) / 256, 256, 0, stream>>>(zbuf, znorm);
  gemm2_kernel<<<dim3(B_DIM / 128, C_PAD / 128), 256, 0, stream>>>(
      zbuf, chi, clo, znorm, cnorm, out);
}
